// BaselineFluxTransformerBlock_9517647528275
// MI455X (gfx1250) — matrix operand order repaired and checked by reading, not compiled, not run on hardware
//
#include <hip/hip_runtime.h>

typedef __bf16 bf16;
typedef __bf16 v16bf __attribute__((ext_vector_type(16)));
typedef float  v8f   __attribute__((ext_vector_type(8)));

#define D_MODEL 1024
#define LN_EPS 1e-5f

__device__ __forceinline__ v8f wmma_bf16(v16bf a, v16bf b, v8f c) {
  // D = A(16x32 bf16) * B(32x16 bf16) + C(16x16 f32)
  return __builtin_amdgcn_wmma_f32_16x16x32_bf16(false, a, false, b, (short)0, c, false, false);
}

// ---------------------------------------------------------------- copy
__global__ void k_copy(const float* __restrict__ s, float* __restrict__ d, int n) {
  int i = blockIdx.x * 256 + threadIdx.x;
  if (i < n) d[i] = s[i];
}

// ---------------------------------------------------------------- layernorm (row = 1024 floats)
__global__ __launch_bounds__(256) void k_layernorm(const float* __restrict__ x,
                                                   const float* __restrict__ g,
                                                   const float* __restrict__ b,
                                                   float* __restrict__ y) {
  __shared__ float red[256];
  const int row = blockIdx.x, t = threadIdx.x;
  const float* xr = x + (size_t)row * D_MODEL;
  float4 v = reinterpret_cast<const float4*>(xr)[t];
  red[t] = v.x + v.y + v.z + v.w;
  __syncthreads();
  for (int o = 128; o > 0; o >>= 1) { if (t < o) red[t] += red[t + o]; __syncthreads(); }
  const float mu = red[0] * (1.0f / D_MODEL);
  __syncthreads();
  float d0 = v.x - mu, d1 = v.y - mu, d2 = v.z - mu, d3 = v.w - mu;
  red[t] = d0 * d0 + d1 * d1 + d2 * d2 + d3 * d3;
  __syncthreads();
  for (int o = 128; o > 0; o >>= 1) { if (t < o) red[t] += red[t + o]; __syncthreads(); }
  const float rstd = rsqrtf(red[0] * (1.0f / D_MODEL) + LN_EPS);
  float* yr = y + (size_t)row * D_MODEL;
  const int c = t * 4;
  yr[c + 0] = d0 * rstd * g[c + 0] + b[c + 0];
  yr[c + 1] = d1 * rstd * g[c + 1] + b[c + 1];
  yr[c + 2] = d2 * rstd * g[c + 2] + b[c + 2];
  yr[c + 3] = d3 * rstd * g[c + 3] + b[c + 3];
}

// ---------------------------------------------------------------- RoPE -> bf16  (x: [S,1024])
__global__ void k_rope_bf16(const float* __restrict__ x, const float* __restrict__ sn,
                            const float* __restrict__ cs, bf16* __restrict__ o) {
  int i = blockIdx.x * 256 + threadIdx.x;        // over S*1024
  int s = i >> 10, c = i & 1023, d = c & 63;
  float xv  = x[i];
  float rot = (d < 32) ? -x[i + 32] : x[i - 32];
  o[i] = (bf16)(xv * cs[s * 64 + d] + rot * sn[s * 64 + d]);
}

// KV buffer [S,2048]: cols 0..1023 = K -> rope -> bf16 [S,1024]
__global__ void k_rope_k(const float* __restrict__ kv, const float* __restrict__ sn,
                         const float* __restrict__ cs, bf16* __restrict__ ko) {
  int i = blockIdx.x * 256 + threadIdx.x;        // over S*1024
  int s = i >> 10, c = i & 1023, d = c & 63;
  const float* row = kv + (size_t)s * 2048;
  float xv  = row[c];
  float rot = (d < 32) ? -row[c + 32] : row[c - 32];
  ko[i] = (bf16)(xv * cs[s * 64 + d] + rot * sn[s * 64 + d]);
}

// V part of KV (cols 1024..2047) -> transposed bf16 V^T [1024, S]
__global__ __launch_bounds__(256) void k_vT(const float* __restrict__ kv,
                                            bf16* __restrict__ vt, int S) {
  __shared__ bf16 tile[32][33];
  const int tx = threadIdx.x, ty = threadIdx.y;   // blockDim = (32,8)
  const int s0 = blockIdx.x * 32, c0 = blockIdx.y * 32;
#pragma unroll
  for (int q = 0; q < 4; ++q) {
    int sl = ty + q * 8;
    tile[sl][tx] = (bf16)kv[(size_t)(s0 + sl) * 2048 + 1024 + c0 + tx];
  }
  __syncthreads();
#pragma unroll
  for (int q = 0; q < 4; ++q) {
    int cl = ty + q * 8;
    vt[(size_t)(c0 + cl) * S + s0 + tx] = tile[tx][cl];
  }
}

// ---------------------------------------------------------------- WMMA GEMM (double-buffered)
// C[M,N] = A[M,K] * W[K,N] + bias ; MODE 0: plain, 1: SiLU, 2: + res
// 256 thr (8 waves), block tile 128x128, wave tile 32x64 (2x4 WMMA), K step 32.
#define AS_STR 40
#define BS_STR 136

template <int MODE>
__global__ __launch_bounds__(256) void k_gemm(const float* __restrict__ A,
                                              const float* __restrict__ W,
                                              const float* __restrict__ bias,
                                              const float* __restrict__ res,
                                              float* __restrict__ C,
                                              int M, int N, int K) {
  __shared__ bf16 As[2][128 * AS_STR];
  __shared__ bf16 Bs[2][32 * BS_STR];
  const int tid = threadIdx.x;
  const int lane = tid & 31, wave = tid >> 5;
  const int wm = wave >> 1, wn = wave & 1;
  const int l15 = lane & 15, half = lane >> 4;
  const int kb8 = half * 8, kb16 = half * 16;
  const int bm = blockIdx.y * 128, bn = blockIdx.x * 128;
  (void)M;

  v8f acc[2][4];
#pragma unroll
  for (int a = 0; a < 2; ++a)
#pragma unroll
    for (int b2 = 0; b2 < 4; ++b2)
#pragma unroll
      for (int r = 0; r < 8; ++r) acc[a][b2][r] = 0.0f;

  // ---- prologue: stage K-tile 0 into buffer 0
  {
#pragma unroll
    for (int it = 0; it < 4; ++it) {
      int idx = tid + it * 256;
      int r = idx >> 3, c4 = (idx & 7) * 4;
      float4 v = *reinterpret_cast<const float4*>(A + (size_t)(bm + r) * K + c4);
      bf16* dst = &As[0][r * AS_STR + c4];
      dst[0] = (bf16)v.x; dst[1] = (bf16)v.y; dst[2] = (bf16)v.z; dst[3] = (bf16)v.w;
    }
#pragma unroll
    for (int it = 0; it < 4; ++it) {
      int idx = tid + it * 256;
      int r = idx >> 5, c4 = (idx & 31) * 4;
      float4 v = *reinterpret_cast<const float4*>(W + (size_t)r * N + bn + c4);
      bf16* dst = &Bs[0][r * BS_STR + c4];
      dst[0] = (bf16)v.x; dst[1] = (bf16)v.y; dst[2] = (bf16)v.z; dst[3] = (bf16)v.w;
    }
  }
  __syncthreads();

  int buf = 0;
  for (int k0 = 0; k0 < K; k0 += 32) {
    const bool more = (k0 + 32) < K;
    // ---- issue next-tile global loads early (in flight during WMMA)
    float4 av[4], bv[4];
    if (more) {
#pragma unroll
      for (int it = 0; it < 4; ++it) {
        int idx = tid + it * 256;
        int r = idx >> 3, c4 = (idx & 7) * 4;
        av[it] = *reinterpret_cast<const float4*>(A + (size_t)(bm + r) * K + (k0 + 32) + c4);
      }
#pragma unroll
      for (int it = 0; it < 4; ++it) {
        int idx = tid + it * 256;
        int r = idx >> 5, c4 = (idx & 31) * 4;
        bv[it] = *reinterpret_cast<const float4*>(W + (size_t)(k0 + 32 + r) * N + bn + c4);
      }
      if (k0 + 64 < K) {   // distance-2 prefetch (CDNA5 global_prefetch)
        __builtin_prefetch(A + (size_t)(bm + (tid >> 3)) * K + (k0 + 64) + (tid & 7) * 4, 0, 3);
        __builtin_prefetch(W + (size_t)(k0 + 64 + (tid >> 5)) * N + bn + (tid & 31) * 4, 0, 3);
      }
    }
    // ---- compute from LDS[buf]
    v16bf af[2], bfr[4];
#pragma unroll
    for (int tm = 0; tm < 2; ++tm) {
      const bf16* ap = &As[buf][(wm * 32 + tm * 16 + l15) * AS_STR];
#pragma unroll
      for (int i = 0; i < 8; ++i) { af[tm][i] = ap[kb8 + i]; af[tm][8 + i] = ap[16 + kb8 + i]; }
    }
#pragma unroll
    for (int tn = 0; tn < 4; ++tn) {
      int col = wn * 64 + tn * 16 + l15;
#pragma unroll
      for (int i = 0; i < 8; ++i) { bfr[tn][i] = Bs[buf][(kb8 + i) * BS_STR + col]; bfr[tn][8 + i] = Bs[buf][(16 + kb8 + i) * BS_STR + col]; }
    }
#pragma unroll
    for (int tm = 0; tm < 2; ++tm)
#pragma unroll
      for (int tn = 0; tn < 4; ++tn)
        acc[tm][tn] = wmma_bf16(af[tm], bfr[tn], acc[tm][tn]);

    // ---- convert + store next tile into the alternate buffer
    if (more) {
#pragma unroll
      for (int it = 0; it < 4; ++it) {
        int idx = tid + it * 256;
        int r = idx >> 3, c4 = (idx & 7) * 4;
        bf16* dst = &As[buf ^ 1][r * AS_STR + c4];
        dst[0] = (bf16)av[it].x; dst[1] = (bf16)av[it].y;
        dst[2] = (bf16)av[it].z; dst[3] = (bf16)av[it].w;
      }
#pragma unroll
      for (int it = 0; it < 4; ++it) {
        int idx = tid + it * 256;
        int r = idx >> 5, c4 = (idx & 31) * 4;
        bf16* dst = &Bs[buf ^ 1][r * BS_STR + c4];
        dst[0] = (bf16)bv[it].x; dst[1] = (bf16)bv[it].y;
        dst[2] = (bf16)bv[it].z; dst[3] = (bf16)bv[it].w;
      }
    }
    __syncthreads();   // one barrier per K-step
    buf ^= 1;
  }

#pragma unroll
  for (int tm = 0; tm < 2; ++tm)
#pragma unroll
    for (int tn = 0; tn < 4; ++tn)
#pragma unroll
      for (int r = 0; r < 8; ++r) {
        int grow = bm + wm * 32 + tm * 16 + r + half * 8;
        int gcol = bn + wn * 64 + tn * 16 + l15;
        float v = acc[tm][tn][r] + bias[gcol];
        if (MODE == 1) v = v / (1.0f + __expf(-v));                 // SiLU
        if (MODE == 2) v += res[(size_t)grow * N + gcol];           // residual
        C[(size_t)grow * N + gcol] = v;
      }
}

// ---------------------------------------------------------------- flash attention (wave32)
// grid: (Sq/128, 16 heads); 8 waves/block, wave -> one 16-row query tile of one head.
// Q,K: bf16 [S,1024] (RoPE'd) ; Vt: bf16 [1024, Sk] (transposed).
__global__ __launch_bounds__(256) void k_attn(const bf16* __restrict__ Q,
                                              const bf16* __restrict__ Kt,
                                              const bf16* __restrict__ Vt,
                                              float* __restrict__ O, int Sk) {
  __shared__ bf16 psh[8][16][40];       // per-wave P (16x32) transpose scratch
  const int tid  = threadIdx.x;
  const int lane = tid & 31, wave = tid >> 5;
  const int l15  = lane & 15, half = lane >> 4;
  const int kb8  = half * 8, kb16 = half * 16;
  const int head = blockIdx.y;
  const int qbase = (blockIdx.x * 8 + wave) * 16;
  const size_t ho = (size_t)head * 64;
  const float scale = 0.125f;           // 1/sqrt(64)

  // Q A-fragments for the two 32-deep chunks of DH=64
  v16bf qa[2];
  {
    const bf16* qr = Q + (size_t)(qbase + l15) * 1024 + ho;
#pragma unroll
    for (int ds = 0; ds < 2; ++ds)
#pragma unroll
      for (int i = 0; i < 8; ++i) {
        qa[ds][i]     = qr[ds * 32 + kb8 + i];
        qa[ds][8 + i] = qr[ds * 32 + 16 + kb8 + i];
      }
  }

  v8f acc[4];
  float mrow[8], lrow[8];
#pragma unroll
  for (int nt = 0; nt < 4; ++nt)
#pragma unroll
    for (int r = 0; r < 8; ++r) acc[nt][r] = 0.0f;
#pragma unroll
  for (int r = 0; r < 8; ++r) { mrow[r] = -1e30f; lrow[r] = 0.0f; }

  for (int j0 = 0; j0 < Sk; j0 += 32) {
    // prefetch next key/value chunk
    if (j0 + 32 < Sk) {
      __builtin_prefetch(Kt + (size_t)(j0 + 32 + l15) * 1024 + ho, 0, 3);
      __builtin_prefetch(Vt + (size_t)(ho + l15) * Sk + j0 + 32, 0, 3);
    }
    // ---- load all four K B-fragments, then WMMA
    v16bf kf[4];
    {
      const bf16* kr0 = Kt + (size_t)(j0 + l15) * 1024 + ho;
      const bf16* kr1 = Kt + (size_t)(j0 + 16 + l15) * 1024 + ho;
#pragma unroll
      for (int i = 0; i < 8; ++i) {
        kf[0][i] = kr0[kb8 + i];      kf[1][i] = kr0[32 + kb8 + i];
        kf[2][i] = kr1[kb8 + i];      kf[3][i] = kr1[32 + kb8 + i];
        kf[0][8 + i] = kr0[16 + kb8 + i];      kf[1][8 + i] = kr0[48 + kb8 + i];
        kf[2][8 + i] = kr1[16 + kb8 + i];      kf[3][8 + i] = kr1[48 + kb8 + i];
      }
    }
    // ---- V fragments issued early: in flight during the softmax VALU work
    // (must be before the asm memory fence below, or they can't be hoisted)
    v16bf vf[4];
#pragma unroll
    for (int nt = 0; nt < 4; ++nt) {
      const bf16* vp = Vt + (size_t)(ho + nt * 16 + l15) * Sk + j0;
#pragma unroll
      for (int i = 0; i < 8; ++i) { vf[nt][i] = vp[kb8 + i]; vf[nt][8 + i] = vp[16 + kb8 + i]; }
    }

    v8f s0 = {0.f,0.f,0.f,0.f,0.f,0.f,0.f,0.f};
    v8f s1 = {0.f,0.f,0.f,0.f,0.f,0.f,0.f,0.f};
    s0 = wmma_bf16(qa[0], kf[0], s0);
    s0 = wmma_bf16(qa[1], kf[1], s0);
    s1 = wmma_bf16(qa[0], kf[2], s1);
    s1 = wmma_bf16(qa[1], kf[3], s1);

    // ---- online softmax over the 32 keys of this block (fully unrolled)
#pragma unroll
    for (int r = 0; r < 8; ++r) {
      float a0 = s0[r] * scale, a1 = s1[r] * scale;
      float rmax = fmaxf(a0, a1);
#pragma unroll
      for (int m = 1; m < 16; m <<= 1) rmax = fmaxf(rmax, __shfl_xor(rmax, m, 32));
      float mnew = fmaxf(mrow[r], rmax);
      float p0 = __expf(a0 - mnew), p1 = __expf(a1 - mnew);
      float rsum = p0 + p1;
#pragma unroll
      for (int m = 1; m < 16; m <<= 1) rsum += __shfl_xor(rsum, m, 32);
      float corr = __expf(mrow[r] - mnew);
      lrow[r] = lrow[r] * corr + rsum;
      mrow[r] = mnew;
#pragma unroll
      for (int nt = 0; nt < 4; ++nt) acc[nt][r] *= corr;
      int prow = r + half * 8;          // C-layout row owned by this lane
      psh[wave][prow][l15]      = (bf16)p0;
      psh[wave][prow][16 + l15] = (bf16)p1;
    }
    asm volatile("s_wait_dscnt 0" ::: "memory");   // wave-local LDS RAW fence

    // ---- O += P(16x32) * V(32x64)
    v16bf pa;
#pragma unroll
    for (int i = 0; i < 8; ++i) {
      pa[i]     = psh[wave][l15][kb8 + i];
      pa[8 + i] = psh[wave][l15][16 + kb8 + i];
    }
#pragma unroll
    for (int nt = 0; nt < 4; ++nt) acc[nt] = wmma_bf16(pa, vf[nt], acc[nt]);
  }

#pragma unroll
  for (int r8 = 0; r8 < 8; ++r8) lrow[r8] = 1.0f / lrow[r8];
#pragma unroll
  for (int nt = 0; nt < 4; ++nt)
#pragma unroll
    for (int r = 0; r < 8; ++r) {
      int qr = qbase + r + half * 8;
      O[(size_t)qr * 1024 + ho + nt * 16 + l15] = acc[nt][r] * lrow[r];
    }
}

// ---------------------------------------------------------------- host launch
extern "C" void kernel_launch(void* const* d_in, const int* in_sizes, int n_in,
                              void* d_out, int out_size, void* d_ws, size_t ws_size,
                              hipStream_t stream) {
  (void)in_sizes; (void)n_in; (void)out_size; (void)ws_size;
  const float* img     = (const float*)d_in[0];
  const float* txt     = (const float*)d_in[1];
  const float* sin_img = (const float*)d_in[2];
  const float* cos_img = (const float*)d_in[3];
  const float* sin_txt = (const float*)d_in[4];
  const float* cos_txt = (const float*)d_in[5];
  const float* sa_q_w  = (const float*)d_in[6];
  const float* sa_q_b  = (const float*)d_in[7];
  const float* sa_kv_w = (const float*)d_in[8];
  const float* sa_kv_b = (const float*)d_in[9];
  const float* sa_o_w  = (const float*)d_in[10];
  const float* sa_o_b  = (const float*)d_in[11];
  const float* ca_q_w  = (const float*)d_in[12];
  const float* ca_q_b  = (const float*)d_in[13];
  const float* ca_kv_w = (const float*)d_in[14];
  const float* ca_kv_b = (const float*)d_in[15];
  const float* ca_o_w  = (const float*)d_in[16];
  const float* ca_o_b  = (const float*)d_in[17];
  const float* ffn1_w  = (const float*)d_in[18];
  const float* ffn1_b  = (const float*)d_in[19];
  const float* ffn2_w  = (const float*)d_in[20];
  const float* ffn2_b  = (const float*)d_in[21];
  const float* ln1_w   = (const float*)d_in[22];
  const float* ln1_b   = (const float*)d_in[23];
  const float* ln2_w   = (const float*)d_in[24];
  const float* ln2_b   = (const float*)d_in[25];
  const float* ln3_w   = (const float*)d_in[26];
  const float* ln3_b   = (const float*)d_in[27];

  float* out_img = (float*)d_out;
  float* out_txt = out_img + (size_t)2048 * 1024;

  char* w = (char*)d_ws;
  float* n1    = (float*)(w);                          // 8 MB  [2048,1024]
  float* qbuf  = (float*)(w + (size_t) 8u * (1u<<20)); // 8 MB
  float* kvbuf = (float*)(w + (size_t)16u * (1u<<20)); // 16 MB [2048,2048]
  bf16*  qr    = (bf16*) (w + (size_t)32u * (1u<<20)); // 4 MB
  bf16*  kr    = (bf16*) (w + (size_t)36u * (1u<<20)); // 4 MB
  bf16*  vt    = (bf16*) (w + (size_t)40u * (1u<<20)); // 4 MB  V^T [1024,2048]
  float* attn  = (float*)(w + (size_t)44u * (1u<<20)); // 8 MB
  float* n2t   = (float*)(w + (size_t)52u * (1u<<20)); // 2 MB  [512,1024]
  float* kvt   = (float*)(w + (size_t)54u * (1u<<20)); // 4 MB  [512,2048]
  bf16*  krt   = (bf16*) (w + (size_t)58u * (1u<<20)); // 1 MB
  bf16*  vtt   = (bf16*) (w + (size_t)59u * (1u<<20)); // 1 MB  V^T [1024,512]
  float* hffn  = (float*)(w + (size_t)60u * (1u<<20)); // 32 MB [2048,4096]

  // residual lives in d_out
  k_copy<<<(2048 * 1024) / 256, 256, 0, stream>>>(img, out_img, 2048 * 1024);
  k_copy<<<(512 * 1024) / 256, 256, 0, stream>>>(txt, out_txt, 512 * 1024);

  // ---- self attention ----
  k_layernorm<<<2048, 256, 0, stream>>>(out_img, ln1_w, ln1_b, n1);
  k_gemm<0><<<dim3(1024 / 128, 2048 / 128), 256, 0, stream>>>(n1, sa_q_w, sa_q_b, nullptr, qbuf, 2048, 1024, 1024);
  k_gemm<0><<<dim3(2048 / 128, 2048 / 128), 256, 0, stream>>>(n1, sa_kv_w, sa_kv_b, nullptr, kvbuf, 2048, 2048, 1024);
  k_rope_bf16<<<(2048 * 1024) / 256, 256, 0, stream>>>(qbuf, sin_img, cos_img, qr);
  k_rope_k<<<(2048 * 1024) / 256, 256, 0, stream>>>(kvbuf, sin_img, cos_img, kr);
  k_vT<<<dim3(2048 / 32, 32), dim3(32, 8), 0, stream>>>(kvbuf, vt, 2048);
  k_attn<<<dim3(2048 / 128, 16), 256, 0, stream>>>(qr, kr, vt, attn, 2048);
  k_gemm<2><<<dim3(1024 / 128, 2048 / 128), 256, 0, stream>>>(attn, sa_o_w, sa_o_b, out_img, out_img, 2048, 1024, 1024);

  // ---- cross attention ----
  k_layernorm<<<2048, 256, 0, stream>>>(out_img, ln2_w, ln2_b, n1);
  k_layernorm<<<512, 256, 0, stream>>>(out_txt, ln2_w, ln2_b, n2t);
  k_gemm<0><<<dim3(1024 / 128, 2048 / 128), 256, 0, stream>>>(n1, ca_q_w, ca_q_b, nullptr, qbuf, 2048, 1024, 1024);
  k_gemm<0><<<dim3(2048 / 128, 512 / 128), 256, 0, stream>>>(n2t, ca_kv_w, ca_kv_b, nullptr, kvt, 512, 2048, 1024);
  k_rope_bf16<<<(2048 * 1024) / 256, 256, 0, stream>>>(qbuf, sin_img, cos_img, qr);
  k_rope_k<<<(512 * 1024) / 256, 256, 0, stream>>>(kvt, sin_txt, cos_txt, krt);
  k_vT<<<dim3(512 / 32, 32), dim3(32, 8), 0, stream>>>(kvt, vtt, 512);
  k_attn<<<dim3(2048 / 128, 16), 256, 0, stream>>>(qr, krt, vtt, attn, 512);
  k_gemm<2><<<dim3(1024 / 128, 2048 / 128), 256, 0, stream>>>(attn, ca_o_w, ca_o_b, out_img, out_img, 2048, 1024, 1024);

  // ---- FFN ----
  k_layernorm<<<2048, 256, 0, stream>>>(out_img, ln3_w, ln3_b, n1);
  k_gemm<1><<<dim3(4096 / 128, 2048 / 128), 256, 0, stream>>>(n1, ffn1_w, ffn1_b, nullptr, hffn, 2048, 4096, 1024);
  k_gemm<2><<<dim3(1024 / 128, 2048 / 128), 256, 0, stream>>>(hffn, ffn2_w, ffn2_b, out_img, out_img, 2048, 1024, 4096);
}
